// ColBERT_38903813767182
// MI455X (gfx1250) — 8-bit matrix operand order repaired and checked by reading, compiled, not run on hardware
//
#include <hip/hip_runtime.h>

#define BB  128
#define LQ  32
#define LD  256
#define HID 768
#define DIM 128

typedef __attribute__((ext_vector_type(16))) __bf16 v16bf;
typedef __attribute__((ext_vector_type(8)))  __bf16 v8bf;
typedef __attribute__((ext_vector_type(8)))  float  v8f;
typedef __attribute__((ext_vector_type(16))) int    v16i;
typedef __attribute__((ext_vector_type(4)))  int    v4i;
typedef __attribute__((ext_vector_type(2)))  int    v2i;

__device__ __forceinline__ __bf16 f2bf(float f) {
  union { float f; unsigned u; } v; v.f = f;
  unsigned r = (v.u + 0x7FFFu + ((v.u >> 16) & 1u)) >> 16;
  union { unsigned short s; __bf16 b; } o; o.s = (unsigned short)r;
  return o.b;
}

__device__ __forceinline__ unsigned char f2fp8(float x) {
  union { float f; unsigned u; } v; v.f = x;
  const unsigned sign = (v.u >> 31) << 7;
  v.u &= 0x7FFFFFFFu;
  const float ax = fminf(v.f, 448.0f);
  v.f = ax;
  const int unbiased = (int)((v.u >> 23) & 0xFFu) - 127;
  const unsigned m = (v.u & 0x7FFFFFu) | 0x800000u;
  const unsigned r = (m + 0x7FFFFu + ((m >> 20) & 1u)) >> 20;
  const unsigned nbits = (unsigned)(((unbiased + 7) << 3) + (int)r - 8);
  const unsigned q = (unsigned)(ax * 512.0f + 0.5f);
  const unsigned bits = (ax < 0.015625f) ? q : nbits;
  return (unsigned char)(sign | bits);
}

__device__ __forceinline__ v16bf load_frag_bf16(const __bf16* __restrict__ p) {
  v8bf lo = *(const v8bf*)(p);
  v8bf hi = *(const v8bf*)(p + 16);
  v16bf f;
#pragma unroll
  for (int i = 0; i < 8; ++i) { f[i] = lo[i]; f[8 + i] = hi[i]; }
  return f;
}

__device__ __forceinline__ v16i load_afrag_fp8(const unsigned char* __restrict__ row,
                                               int half) {
  v16i f;
#pragma unroll
  for (int g = 0; g < 8; ++g) {
    v2i c = *(const v2i*)(row + 16 * g + 8 * half);
    f[2 * g] = c[0]; f[2 * g + 1] = c[1];
  }
  return f;
}

__device__ __forceinline__ v16i load_bfrag_fp8(const unsigned char* __restrict__ col,
                                               int half) {
  v16i f;
#pragma unroll
  for (int q = 0; q < 8; ++q) {
    v2i c = *(const v2i*)(col + 16 * q + 8 * half);
    f[2 * q] = c[0]; f[2 * q + 1] = c[1];
  }
  return f;
}

__global__ __launch_bounds__(256) void proj_l2_kernel(
    const float* __restrict__ hidden,
    const float* __restrict__ W,
    const int*   __restrict__ mask,
    unsigned char* __restrict__ outp)
{
  __shared__ __align__(16) __bf16 Wlds[DIM * 32];

  const int wave = threadIdx.x >> 5;
  const int lane = threadIdx.x & 31;
  const int r    = lane & 15;
  const int half = lane >> 4;
  const int kb   = half * 8;
  const int row0 = (blockIdx.x * 8 + wave) * 16;
  const int arow = row0 + r;

  v8f acc[8];
#pragma unroll
  for (int t = 0; t < 8; ++t)
#pragma unroll
    for (int j = 0; j < 8; ++j) acc[t][j] = 0.f;

  for (int k0 = 0; k0 < HID; k0 += 32) {
    __syncthreads();
    for (int e = threadIdx.x; e < DIM * 32; e += 256) {
      int d = e >> 5, kk = e & 31;
      Wlds[e] = f2bf(W[(size_t)d * HID + k0 + kk]);
    }
    __syncthreads();

    const float* ap = hidden + (size_t)arow * HID + k0 + kb;
    v8f alo = *(const v8f*)(ap);
    v8f ahi = *(const v8f*)(ap + 16);
    v16bf afrag;
#pragma unroll
    for (int i = 0; i < 8; ++i) { afrag[i] = f2bf(alo[i]); afrag[8 + i] = f2bf(ahi[i]); }

#pragma unroll
    for (int t = 0; t < 8; ++t) {
      v16bf bfrag = load_frag_bf16(Wlds + ((t * 16 + r) * 32 + kb));
      acc[t] = __builtin_amdgcn_wmma_f32_16x16x32_bf16(
          false, afrag, false, bfrag, (short)0, acc[t], false, false);
    }
  }

#pragma unroll
  for (int j = 0; j < 8; ++j) {
    float ss = 0.f;
#pragma unroll
    for (int t = 0; t < 8; ++t) ss += acc[t][j] * acc[t][j];
#pragma unroll
    for (int m = 1; m < 16; m <<= 1) ss += __shfl_xor(ss, m, 32);
    const int row = row0 + half * 8 + j;
    float scale = rsqrtf(ss);
    if (mask) scale *= (mask[row] != 0) ? 1.f : 0.f;
#pragma unroll
    for (int t = 0; t < 8; ++t)
      outp[(size_t)row * DIM + t * 16 + r] = f2fp8(acc[t][j] * scale);
  }
}

__global__ __launch_bounds__(256) void maxsim_kernel(
    const unsigned char* __restrict__ Qp,
    const unsigned char* __restrict__ Dp,
    const int*           __restrict__ mask,
    float*               __restrict__ out)
{
  const int c    = blockIdx.x;
  const int b    = blockIdx.y;
  const int wave = threadIdx.x >> 5;
  const int lane = threadIdx.x & 31;
  const int r    = lane & 15;
  const int half = lane >> 4;
  const int n0   = wave * 32;

  __shared__ float redbuf[8][32];

  const unsigned char* qb = Qp + (size_t)b * LQ * DIM;
  const unsigned char* db = Dp + (size_t)c * LD * DIM;

  v16i a0 = load_afrag_fp8(qb + (size_t)(r)      * DIM, half);
  v16i a1 = load_afrag_fp8(qb + (size_t)(16 + r) * DIM, half);
  v16i b0 = load_bfrag_fp8(db + (size_t)(n0 + r)      * DIM, half);
  v16i b1 = load_bfrag_fp8(db + (size_t)(n0 + 16 + r) * DIM, half);

  v8f z;
#pragma unroll
  for (int j = 0; j < 8; ++j) z[j] = 0.f;

  v8f acc[2][2];
  acc[0][0] = __builtin_amdgcn_wmma_f32_16x16x128_fp8_fp8(a0, b0, (short)0, z, false, false);
  acc[0][1] = __builtin_amdgcn_wmma_f32_16x16x128_fp8_fp8(a0, b1, (short)0, z, false, false);
  acc[1][0] = __builtin_amdgcn_wmma_f32_16x16x128_fp8_fp8(a1, b0, (short)0, z, false, false);
  acc[1][1] = __builtin_amdgcn_wmma_f32_16x16x128_fp8_fp8(a1, b1, (short)0, z, false, false);

  const float NEG = -3.0e38f;
  const int mv0 = mask[(size_t)c * LD + n0 + r];
  const int mv1 = mask[(size_t)c * LD + n0 + 16 + r];

#pragma unroll
  for (int mt = 0; mt < 2; ++mt) {
#pragma unroll
    for (int j = 0; j < 8; ++j) {
      float v = fmaxf(mv0 ? acc[mt][0][j] : NEG, mv1 ? acc[mt][1][j] : NEG);
#pragma unroll
      for (int m = 1; m < 16; m <<= 1) v = fmaxf(v, __shfl_xor(v, m, 32));
      if (r == 0) redbuf[wave][mt * 16 + half * 8 + j] = v;
    }
  }
  __syncthreads();

  if (threadIdx.x < 32) {
    float v = redbuf[0][threadIdx.x];
#pragma unroll
    for (int w = 1; w < 8; ++w) v = fmaxf(v, redbuf[w][threadIdx.x]);
#pragma unroll
    for (int m = 1; m < 32; m <<= 1) v += __shfl_xor(v, m, 32);
    if (threadIdx.x == 0) out[(size_t)b * BB + c] = v;
  }
}

extern "C" void kernel_launch(void* const* d_in, const int* in_sizes, int n_in,
                              void* d_out, int out_size, void* d_ws, size_t ws_size,
                              hipStream_t stream) {
  (void)in_sizes; (void)n_in; (void)out_size; (void)ws_size;
  const float* q_hidden = (const float*)d_in[0];
  const float* d_hidden = (const float*)d_in[1];
  const float* W        = (const float*)d_in[2];
  const int*   d_mask   = (const int*)d_in[3];

  unsigned char* Qp = (unsigned char*)d_ws;
  unsigned char* Dp = Qp + (size_t)BB * LQ * DIM;

  proj_l2_kernel<<<dim3((BB * LQ) / (16 * 8)), 256, 0, stream>>>(q_hidden, W, nullptr, Qp);
  proj_l2_kernel<<<dim3((BB * LD) / (16 * 8)), 256, 0, stream>>>(d_hidden, W, d_mask, Dp);
  maxsim_kernel<<<dim3(BB, BB), 256, 0, stream>>>(Qp, Dp, d_mask, (float*)d_out);
}
